// EdgeBlockLite_86844238725707
// MI455X (gfx1250) — matrix operand order repaired and checked by reading, not compiled, not run on hardware
//
#include <hip/hip_runtime.h>
#include <hip/hip_bf16.h>

typedef __attribute__((ext_vector_type(16))) _Float16 v16h;
typedef __attribute__((ext_vector_type(8)))  _Float16 v8h;
typedef __attribute__((ext_vector_type(8)))  float    v8f;
typedef __attribute__((ext_vector_type(4)))  int      v4i;

#if __has_builtin(__builtin_amdgcn_global_load_async_to_lds_b128)
#define HAVE_ASYNC_LOAD 1
#else
#define HAVE_ASYNC_LOAD 0
#endif
#if __has_builtin(__builtin_amdgcn_global_store_async_from_lds_b128)
#define HAVE_ASYNC_STORE 1
#else
#define HAVE_ASYNC_STORE 0
#endif

typedef __attribute__((address_space(1))) char  glb_char;
typedef __attribute__((address_space(3))) char  lds_char;
typedef __attribute__((address_space(1))) v4i   glb_v4i;   // AS1 ("__device__") per diagnostic
typedef __attribute__((address_space(3))) v4i   lds_v4i;   // AS3 ("__shared__")

__device__ __forceinline__ void wait_asynccnt0() {
#if __has_builtin(__builtin_amdgcn_s_wait_asynccnt)
  __builtin_amdgcn_s_wait_asynccnt(0);
#else
  asm volatile("s_wait_asynccnt 0" ::: "memory");
#endif
}

namespace {
constexpr int Bg = 64, Nn = 128, Kk = 32;
constexpr int E_ = Bg * Nn * Kk;          // 262144 edges
constexpr int NDIM = 128, EDIM = 128, GDIM = 64;
constexpr int EDGE_IN = EDIM + 2 * NDIM;  // 384 (LayerNorm width)
constexpr int XIN = EDGE_IN + GDIM;       // 448 (MLP input width)
constexpr int HID = 256, HID2 = 512, OUT = 128, HEADS = 4;
constexpr int LCOLS = 16;                 // attn_w2 padded to 16 columns for WMMA
constexpr int MT = 64;                    // edges per block (= 2 softmax segments)

// LDS strides (halves), padded to stay 16B aligned and spread banks
constexpr int XS = 456;                   // xin row stride (448 + 8)
constexpr int HS = 520;                   // hidden row stride (512 + 8)
constexpr int OFF_XIN = 0;                              // also edges_out overlay
constexpr int OFF_HID = MT * XS * 2;                    // 58368
constexpr int OFF_EFS = OFF_HID + MT * HS * 2;          // 124928
constexpr int OFF_WLG = OFF_EFS + MT * EDIM * 4;        // 157696
constexpr int SMEM_BYTES = OFF_WLG + MT * HEADS * 4;    // 158720 (< 320KB WGP LDS)

// workspace layout (bytes)
constexpr size_t WS_W1T = 0;                                    // 512*448 f16
constexpr size_t WS_W2T = WS_W1T + (size_t)HID2 * XIN * 2;      // 128*256 f16
constexpr size_t WS_W2A = WS_W2T + (size_t)OUT * HID * 2;       // 16*256 f16 (attn_w2 padded)
}

// ---------------------------------------------------------------------------
// Kernel 0: convert + transpose weights to f16 once per launch (into d_ws).
// w1t[c][k]: c in [0,512): 0-255 = attn_w1 cols, 256-511 = feat_w1 cols.
// w2t[c][k]: feat_w2 transposed. w2a[c][k]: attn_w2 transposed, zero-padded
// from 4 to 16 columns so logits ride the WMMA pipe.
// Column-major-by-K so a B-fragment is one contiguous 32B load per lane.
// ---------------------------------------------------------------------------
__global__ void prep_weights(const float* __restrict__ aw1,
                             const float* __restrict__ fw1,
                             const float* __restrict__ fw2,
                             const float* __restrict__ aw2,
                             _Float16* __restrict__ w1t,
                             _Float16* __restrict__ w2t,
                             _Float16* __restrict__ w2a) {
  int idx = blockIdx.x * blockDim.x + threadIdx.x;
  if (idx < XIN * HID2) {                 // 229376
    int c = idx / XIN, k = idx % XIN;
    float v = (c < HID) ? aw1[k * HID + c] : fw1[k * HID + (c - HID)];
    w1t[c * XIN + k] = (_Float16)v;
  }
  if (idx < HID * OUT) {                  // 32768
    int c = idx / HID, k = idx % HID;
    w2t[c * HID + k] = (_Float16)fw2[k * OUT + c];
  }
  if (idx < HID * LCOLS) {                // 4096
    int c = idx / HID, k = idx % HID;
    w2a[c * HID + k] = (c < HEADS) ? (_Float16)aw2[k * HEADS + c] : (_Float16)0.f;
  }
}

// ---------------------------------------------------------------------------
// Main fused kernel: gather + LN + ctxt -> GEMM1 (WMMA, fused attn|feat) ->
// SiLU -> GEMM2 + logits (WMMA) + residual -> segment softmax -> pooling.
// One block handles 64 edges = 2 complete receiver segments (no atomics).
// ---------------------------------------------------------------------------
__global__ __launch_bounds__(256)
void edge_block(const float* __restrict__ nodes,
                const float* __restrict__ edge_feats,
                const float* __restrict__ globs,
                const float* __restrict__ ln_gamma,
                const float* __restrict__ ln_beta,
                const float* __restrict__ attn_b1,
                const float* __restrict__ attn_b2,
                const float* __restrict__ feat_b1,
                const float* __restrict__ feat_b2,
                const int* __restrict__ edge_b,
                const int* __restrict__ edge_i,
                const int* __restrict__ edge_j,
                const _Float16* __restrict__ w1t,
                const _Float16* __restrict__ w2t,
                const _Float16* __restrict__ w2a,
                float* __restrict__ out_edges,
                float* __restrict__ out_pooled) {
  extern __shared__ char smem[];
  _Float16* xin = (_Float16*)(smem + OFF_XIN);
  _Float16* hid = (_Float16*)(smem + OFF_HID);
  float*    efs = (float*)(smem + OFF_EFS);
  float*    wlg = (float*)(smem + OFF_WLG);
  float*    eo  = (float*)(smem + OFF_XIN);   // overlays xin after GEMM1

  const int tid  = threadIdx.x;
  const int lane = tid & 31;
  const int w    = tid >> 5;                  // wave id, 8 waves
  const int e0   = blockIdx.x * MT;

  // CDNA5 global prefetch of the streamed f16 weights (global_prefetch_b8)
  __builtin_prefetch(w1t + tid * 64, 0, 0);
  __builtin_prefetch(w2t + tid * 64, 0, 0);

  // ---- Stage edge_feats tile into LDS (used by LN and by the residual) ----
#if HAVE_ASYNC_LOAD
  {
    glb_v4i* g = (glb_v4i*)((glb_char*)(edge_feats + (size_t)e0 * EDIM) + tid * 16);
    lds_v4i* l = (lds_v4i*)((lds_char*)(&smem[OFF_EFS]) + tid * 16);
#pragma unroll
    for (int it = 0; it < (MT * EDIM * 4) / (256 * 16); ++it)   // 8 x b128 per thread
      __builtin_amdgcn_global_load_async_to_lds_b128(g + it * 256, l + it * 256, 0, 0);
    wait_asynccnt0();
  }
#else
  {
    const float4* src = (const float4*)(edge_feats + (size_t)e0 * EDIM);
    float4* dst = (float4*)efs;
#pragma unroll
    for (int it = 0; it < (MT * EDIM / 4) / 256; ++it)
      dst[tid + it * 256] = src[tid + it * 256];
  }
#endif
  __syncthreads();

  // ---- Gather + LayerNorm + ctxt: one wave per edge row, f32 math ----
  for (int rr = 0; rr < MT / 8; ++rr) {
    const int r = w + 8 * rr;
    const int e = e0 + r;
    const int b = edge_b[e], si = edge_i[e], sj = edge_j[e];
    const float* ps = nodes + ((size_t)b * Nn + si) * NDIM;
    const float* pr = nodes + ((size_t)b * Nn + sj) * NDIM;
    float x[12];
#pragma unroll
    for (int k = 0; k < 4; ++k) x[k]     = ps[lane + 32 * k];
#pragma unroll
    for (int k = 0; k < 4; ++k) x[4 + k] = pr[lane + 32 * k];
#pragma unroll
    for (int k = 0; k < 4; ++k) x[8 + k] = efs[r * EDIM + lane + 32 * k];
    float s = 0.f, s2 = 0.f;
#pragma unroll
    for (int k = 0; k < 12; ++k) { s += x[k]; s2 += x[k] * x[k]; }
#pragma unroll
    for (int off = 16; off; off >>= 1) {     // wave32 tree reduction
      s  += __shfl_xor(s,  off, 32);
      s2 += __shfl_xor(s2, off, 32);
    }
    const float mu  = s * (1.f / EDGE_IN);
    const float var = s2 * (1.f / EDGE_IN) - mu * mu;
    const float inv = rsqrtf(var + 1e-5f);
#pragma unroll
    for (int k = 0; k < 12; ++k) {
      const int c = lane + 32 * k;
      xin[r * XS + c] = (_Float16)((x[k] - mu) * inv * ln_gamma[c] + ln_beta[c]);
    }
    xin[r * XS + EDGE_IN + lane]      = (_Float16)globs[b * GDIM + lane];
    xin[r * XS + EDGE_IN + 32 + lane] = (_Float16)globs[b * GDIM + 32 + lane];
  }
  __syncthreads();

  // WMMA geometry: wave -> (row tile, column half)
  const int rt   = w >> 1;          // 0..3 -> rows 16*rt..16*rt+15
  const int row0 = rt * 16;
  const int ch   = w & 1;           // 0 = attn half, 1 = feat half of hidden
  const int m    = lane & 15;       // A: row / B,C: column within tile
  const int hi   = lane >> 4;       // K/M-half selector per ISA layouts

  // ---- GEMM1: hidden[64 x 512] = xin[64 x 448] @ w1t^T, SiLU fused ----
  for (int cc = 0; cc < 4; ++cc) {
    const int colb = ch * HID + cc * 64;
    v8f acc0 = {}, acc1 = {}, acc2 = {}, acc3 = {};
    for (int kt = 0; kt < XIN / 32; ++kt) {
      const int k0 = kt * 32;
      union { v16h v; v8h h[2]; } a;
      a.h[0] = *(const v8h*)&xin[(row0 + m) * XS + k0 + hi * 16];
      a.h[1] = *(const v8h*)&xin[(row0 + m) * XS + k0 + hi * 16 + 8];
      const _Float16* bp = w1t + (size_t)(colb + m) * XIN + k0 + hi * 16;
      v16h b0 = *(const v16h*)(bp);
      v16h b1 = *(const v16h*)(bp + 16 * XIN);
      v16h b2 = *(const v16h*)(bp + 32 * XIN);
      v16h b3 = *(const v16h*)(bp + 48 * XIN);
      acc0 = __builtin_amdgcn_wmma_f32_16x16x32_f16(false, a.v, false, b0, (short)0, acc0, false, false);
      acc1 = __builtin_amdgcn_wmma_f32_16x16x32_f16(false, a.v, false, b1, (short)0, acc1, false, false);
      acc2 = __builtin_amdgcn_wmma_f32_16x16x32_f16(false, a.v, false, b2, (short)0, acc2, false, false);
      acc3 = __builtin_amdgcn_wmma_f32_16x16x32_f16(false, a.v, false, b3, (short)0, acc3, false, false);
    }
    auto store_h = [&](v8f c, int t) {
      const int col = colb + t * 16 + m;
      const float b1v = (col < HID) ? attn_b1[col] : feat_b1[col - HID];
#pragma unroll
      for (int i = 0; i < 8; ++i) {         // C layout: M = i + 8*hi, N = m
        const int row = row0 + i + 8 * hi;
        float h = c[i] + b1v;
        h = h / (1.f + __expf(-h));          // SiLU
        hid[row * HS + col] = (_Float16)h;
      }
    };
    store_h(acc0, 0); store_h(acc1, 1); store_h(acc2, 2); store_h(acc3, 3);
  }
  __syncthreads();

  // ---- GEMM2: edges_out[64x128] = silu_feat[64x256] @ w2t^T (+bias+residual)
  //      ch==0 waves also fold the attention logits [64x16-padded] in as one
  //      extra WMMA tile per k-step (A = silu_attn half, B = padded attn_w2).
  {
    const int colb2 = ch * 64;
    v8f acc0 = {}, acc1 = {}, acc2 = {}, acc3 = {}, accL = {};
    for (int kt = 0; kt < HID / 32; ++kt) {
      const int k0 = kt * 32;
      union { v16h v; v8h h[2]; } a;
      a.h[0] = *(const v8h*)&hid[(row0 + m) * HS + HID + k0 + hi * 16];
      a.h[1] = *(const v8h*)&hid[(row0 + m) * HS + HID + k0 + hi * 16 + 8];
      const _Float16* bp = w2t + (size_t)(colb2 + m) * HID + k0 + hi * 16;
      v16h b0 = *(const v16h*)(bp);
      v16h b1 = *(const v16h*)(bp + 16 * HID);
      v16h b2 = *(const v16h*)(bp + 32 * HID);
      v16h b3 = *(const v16h*)(bp + 48 * HID);
      acc0 = __builtin_amdgcn_wmma_f32_16x16x32_f16(false, a.v, false, b0, (short)0, acc0, false, false);
      acc1 = __builtin_amdgcn_wmma_f32_16x16x32_f16(false, a.v, false, b1, (short)0, acc1, false, false);
      acc2 = __builtin_amdgcn_wmma_f32_16x16x32_f16(false, a.v, false, b2, (short)0, acc2, false, false);
      acc3 = __builtin_amdgcn_wmma_f32_16x16x32_f16(false, a.v, false, b3, (short)0, acc3, false, false);
      if (ch == 0) {                         // wave-uniform: EXEC stays all-1s
        union { v16h v; v8h h[2]; } aa;
        aa.h[0] = *(const v8h*)&hid[(row0 + m) * HS + k0 + hi * 16];
        aa.h[1] = *(const v8h*)&hid[(row0 + m) * HS + k0 + hi * 16 + 8];
        v16h bL = *(const v16h*)&w2a[(size_t)m * HID + k0 + hi * 16];
        accL = __builtin_amdgcn_wmma_f32_16x16x32_f16(false, aa.v, false, bL, (short)0, accL, false, false);
      }
    }
    auto store_o = [&](v8f c, int t) {
      const int col = colb2 + t * 16 + m;
      const float b2v = feat_b2[col];
#pragma unroll
      for (int i = 0; i < 8; ++i) {
        const int row = row0 + i + 8 * hi;
        eo[row * OUT + col] = c[i] + b2v + efs[row * EDIM + col];  // + residual
      }
    };
    store_o(acc0, 0); store_o(acc1, 1); store_o(acc2, 2); store_o(acc3, 3);
    if (ch == 0 && m < HEADS) {              // logits: only cols 0..3 are real
      const float b2a = attn_b2[m];
#pragma unroll
      for (int i = 0; i < 8; ++i)
        wlg[(row0 + i + 8 * hi) * HEADS + m] = accL[i] + b2a;
    }
  }
  __syncthreads();

  // ---- Write edges_out tile back to global, coalesced from LDS.
  //      Async store overlaps with the softmax below; s_endpgm's implicit
  //      wait-idle guarantees completion.
#if HAVE_ASYNC_STORE
  {
    glb_v4i* g = (glb_v4i*)((glb_char*)(out_edges + (size_t)e0 * OUT) + tid * 16);
    lds_v4i* l = (lds_v4i*)((lds_char*)(&smem[OFF_XIN]) + tid * 16);
#pragma unroll
    for (int it = 0; it < (MT * OUT * 4) / (256 * 16); ++it)   // 8 x b128 per thread
      __builtin_amdgcn_global_store_async_from_lds_b128(g + it * 256, l + it * 256, 0, 0);
  }
#else
  {
    const float4* src = (const float4*)eo;
    float4* dst = (float4*)(out_edges + (size_t)e0 * OUT);
#pragma unroll
    for (int it = 0; it < (MT * OUT / 4) / 256; ++it)
      dst[tid + it * 256] = src[tid + it * 256];
  }
#endif

  // ---- Segment softmax over the 2 local segments (K=32 contiguous edges) ----
  if (tid < 2 * HEADS) {
    const int s = tid >> 2, h = tid & 3;
    float mx = -3.4e38f;
    for (int r = 0; r < Kk; ++r) mx = fmaxf(mx, wlg[(s * Kk + r) * HEADS + h]);
    float den = 0.f;
    for (int r = 0; r < Kk; ++r) den += __expf(wlg[(s * Kk + r) * HEADS + h] - mx);
    const float inv = 1.f / den;
    for (int r = 0; r < Kk; ++r) {
      const float l = wlg[(s * Kk + r) * HEADS + h];
      wlg[(s * Kk + r) * HEADS + h] = __expf(l - mx) * inv;
    }
  }
  __syncthreads();

  // ---- Weighted pool to receivers (block-local: no atomics) ----
  {
    const int s = tid >> 7;          // segment 0/1
    const int c = tid & 127;         // output channel
    float acc = 0.f;
#pragma unroll 4
    for (int r = 0; r < Kk; ++r)
      acc += eo[(s * Kk + r) * OUT + c] * wlg[(s * Kk + r) * HEADS + (c >> 5)];
    out_pooled[(size_t)(blockIdx.x * 2 + s) * OUT + c] = acc * 0.08838834764831845f; // /sqrt(128)
  }
}

// ---------------------------------------------------------------------------
extern "C" void kernel_launch(void* const* d_in, const int* in_sizes, int n_in,
                              void* d_out, int out_size, void* d_ws, size_t ws_size,
                              hipStream_t stream) {
  const float* nodes      = (const float*)d_in[0];
  const float* edge_feats = (const float*)d_in[1];
  const float* globs      = (const float*)d_in[2];
  const float* ln_gamma   = (const float*)d_in[3];
  const float* ln_beta    = (const float*)d_in[4];
  const float* attn_w1    = (const float*)d_in[5];
  const float* attn_b1    = (const float*)d_in[6];
  const float* attn_w2    = (const float*)d_in[7];
  const float* attn_b2    = (const float*)d_in[8];
  const float* feat_w1    = (const float*)d_in[9];
  const float* feat_b1    = (const float*)d_in[10];
  const float* feat_w2    = (const float*)d_in[11];
  const float* feat_b2    = (const float*)d_in[12];
  const int*   edge_b     = (const int*)d_in[13];
  const int*   edge_i     = (const int*)d_in[14];
  const int*   edge_j     = (const int*)d_in[15];

  _Float16* w1t = (_Float16*)((char*)d_ws + WS_W1T);
  _Float16* w2t = (_Float16*)((char*)d_ws + WS_W2T);
  _Float16* w2a = (_Float16*)((char*)d_ws + WS_W2A);

  float* out_edges  = (float*)d_out;
  float* out_pooled = out_edges + (size_t)E_ * OUT;

  (void)hipFuncSetAttribute((const void*)edge_block,
                            hipFuncAttributeMaxDynamicSharedMemorySize, SMEM_BYTES);

  prep_weights<<<(XIN * HID2 + 255) / 256, 256, 0, stream>>>(
      attn_w1, feat_w1, feat_w2, attn_w2, w1t, w2t, w2a);

  edge_block<<<E_ / MT, 256, SMEM_BYTES, stream>>>(
      nodes, edge_feats, globs, ln_gamma, ln_beta,
      attn_b1, attn_b2, feat_b1, feat_b2,
      edge_b, edge_i, edge_j, w1t, w2t, w2a, out_edges, out_pooled);
}
